// GAIN_8194797601165
// MI455X (gfx1250) — matrix operand order repaired and checked by reading, not compiled, not run on hardware
//
#include <hip/hip_runtime.h>
#include <hip/hip_bf16.h>
#include <math.h>

// ---------------------------------------------------------------------------
// GATv2 (4 layers) for MI455X / gfx1250.
// GEMMs use v_wmma_f32_16x16x32_f16 with split-f16 (hi/lo) error correction:
//   A*B ~= Ahi*Bhi + Ahi*Blo + Alo*Bhi   (f32 accumulate)  -> ~f32 accuracy.
// Each wave computes a 16x64 output tile (4x N-blocking) so the A fragment is
// reused across 4 WMMA column tiles: 12 WMMAs per 20 b128 loads per K-step.
// Graph softmax/aggregation uses float atomics (max via monotone-uint trick).
// ---------------------------------------------------------------------------

#define N_NODES 50000
#define N_EDGES 200000
#define NEG_SLOPE 0.2f

typedef __attribute__((ext_vector_type(16))) _Float16 v16h;
typedef __attribute__((ext_vector_type(8)))  _Float16 v8h;
typedef __attribute__((ext_vector_type(8)))  float    v8f;

union V16U { v16h v; v8h h[2]; };

// ----- monotone order-preserving float<->uint encoding (for atomicMax) -----
__device__ __forceinline__ unsigned fenc(float f) {
  unsigned u = __float_as_uint(f);
  return (u & 0x80000000u) ? ~u : (u | 0x80000000u);
}
__device__ __forceinline__ float fdec(unsigned e) {
  return (e & 0x80000000u) ? __uint_as_float(e ^ 0x80000000u)
                           : __uint_as_float(~e);
}
#define ENC_NEG_INF 0x007FFFFFu  // fenc(-inf)

// ---------------------------------------------------------------------------
// Utility: fill u32 pattern (zeros for f32 buffers, ENC_NEG_INF for max bufs)
// ---------------------------------------------------------------------------
__global__ void fill_u32(unsigned* __restrict__ p, unsigned v, long long n) {
  long long i = (long long)blockIdx.x * blockDim.x + threadIdx.x;
  if (i < n) p[i] = v;
}

// ---------------------------------------------------------------------------
// f32 -> (hi f16, lo f16) split
// ---------------------------------------------------------------------------
__global__ void split_f16(const float* __restrict__ X,
                          _Float16* __restrict__ hi, _Float16* __restrict__ lo,
                          long long n) {
  long long i = (long long)blockIdx.x * blockDim.x + threadIdx.x;
  if (i >= n) return;
  float v = X[i];
  _Float16 h = (_Float16)v;
  hi[i] = h;
  lo[i] = (_Float16)(v - (float)h);
}

// W [K, Nout] row-major  ->  Wt hi/lo [Nout, K] (transposed, split)
__global__ void transpose_split_f16(const float* __restrict__ W,
                                    _Float16* __restrict__ hi,
                                    _Float16* __restrict__ lo,
                                    int K, int Nout) {
  long long i = (long long)blockIdx.x * blockDim.x + threadIdx.x;
  if (i >= (long long)K * Nout) return;
  int j = (int)(i / K);           // output column (row of Wt)
  int k = (int)(i % K);
  float v = W[(long long)k * Nout + j];
  _Float16 h = (_Float16)v;
  hi[i] = h;
  lo[i] = (_Float16)(v - (float)h);
}

// ---------------------------------------------------------------------------
// GEMM:  Y[M,Nout] = A[M,K] * W[K,Nout] + bias   (A,W pre-split into f16 hi/lo,
// W pre-transposed to [Nout,K]).  One wave computes a 16x64 output tile
// (4 x 16x16 WMMA tiles sharing one A fragment per K-step).
// A fragment (16x32 f16): lane<16 -> M=lane, K {kk..kk+15}
//                         lane>=16 -> M=lane-16, K {kk+16..kk+31}
// B fragment (32x16 f16): lane -> N=lane&15, 16 consecutive K starting at
//                         kk + (lane<16 ? 0 : 16)   (from transposed W rows)
// ---------------------------------------------------------------------------
__global__ __launch_bounds__(256)
void gemm_split_wmma(const _Float16* __restrict__ Ahi, const _Float16* __restrict__ Alo,
                     const _Float16* __restrict__ Bhi, const _Float16* __restrict__ Blo,
                     const float* __restrict__ bias, float* __restrict__ Y,
                     int M, int K, int Nout) {
  int wid  = blockIdx.x * (blockDim.x >> 5) + (threadIdx.x >> 5);
  int lane = threadIdx.x & 31;
  int tilesN = Nout >> 6;                 // 64-wide N tiles
  int tM = (wid / tilesN) << 4;
  int tN = (wid % tilesN) << 6;
  if (tM >= M) return;

  int hi16 = lane >> 4;                   // 0 | 1
  int l15  = lane & 15;
  long long rowA = (long long)(tM + l15) * K;
  int aoff = hi16 << 4;                   // 0 | 16 (halfs)
  int boff = hi16 << 4;                   // 0 | 16 (halfs)

  const _Float16* pBh = Bhi + (long long)(tN + l15) * K + boff;
  const _Float16* pBl = Blo + (long long)(tN + l15) * K + boff;
  const long long colStride = (long long)16 * K;   // next 16-col tile of Wt

  v8f c[4] = {{}, {}, {}, {}};
  for (int kk = 0; kk < K; kk += 32) {
    const v8h* pAh = (const v8h*)(Ahi + rowA + kk + aoff);
    const v8h* pAl = (const v8h*)(Alo + rowA + kk + aoff);
    if (kk + 32 < K) {                    // prefetch A two K-steps ahead (kk+64)
      __builtin_prefetch((const char*)(pAh + 8), 0, 1);
      __builtin_prefetch((const char*)(pAl + 8), 0, 1);
    }
    V16U ahi, alo;
    ahi.h[0] = pAh[0]; ahi.h[1] = pAh[1];   // +8 halfs
    alo.h[0] = pAl[0]; alo.h[1] = pAl[1];
#pragma unroll
    for (int j = 0; j < 4; ++j) {
      v16h bh = *(const v16h*)(pBh + j * colStride + kk);
      v16h bl = *(const v16h*)(pBl + j * colStride + kk);
      c[j] = __builtin_amdgcn_wmma_f32_16x16x32_f16(false, ahi.v, false, bh,
                                                    (short)0, c[j], false, false);
      c[j] = __builtin_amdgcn_wmma_f32_16x16x32_f16(false, ahi.v, false, bl,
                                                    (short)0, c[j], false, false);
      c[j] = __builtin_amdgcn_wmma_f32_16x16x32_f16(false, alo.v, false, bh,
                                                    (short)0, c[j], false, false);
    }
  }

  // C/D layout: lane<16 -> VGPR g is M=g ; lane>=16 -> M=g+8 ; N = lane&15
  int mbase = tM + (hi16 << 3);
#pragma unroll
  for (int j = 0; j < 4; ++j) {
    int col = tN + (j << 4) + l15;
    float bv = bias[col];
#pragma unroll
    for (int g = 0; g < 8; ++g)
      Y[(long long)(mbase + g) * Nout + col] = c[j][g] + bv;
  }
}

// ---------------------------------------------------------------------------
// Edge pass 1: e[e,h] = att[h] . leaky_relu(xl[src] + xr[dst]) ; atomic max
// ---------------------------------------------------------------------------
__global__ void edge_score(const float* __restrict__ XL, const float* __restrict__ XR,
                           const long long* __restrict__ src, const long long* __restrict__ dst,
                           const float* __restrict__ att,
                           float* __restrict__ eOut, unsigned* __restrict__ maxEnc,
                           int E, int H, int C) {
  long long t = (long long)blockIdx.x * blockDim.x + threadIdx.x;
  if (t >= (long long)E * H) return;
  int e = (int)(t / H), h = (int)(t % H);
  long long s = src[e], d = dst[e];
  const float4* pl = (const float4*)(XL + ((long long)s * H + h) * C);
  const float4* pr = (const float4*)(XR + ((long long)d * H + h) * C);
  const float4* pa = (const float4*)(att + (long long)h * C);
  float acc = 0.f;
  for (int i = 0; i < (C >> 2); ++i) {
    float4 l = pl[i], r = pr[i], a = pa[i];
    float z0 = l.x + r.x; z0 = z0 > 0.f ? z0 : NEG_SLOPE * z0;
    float z1 = l.y + r.y; z1 = z1 > 0.f ? z1 : NEG_SLOPE * z1;
    float z2 = l.z + r.z; z2 = z2 > 0.f ? z2 : NEG_SLOPE * z2;
    float z3 = l.w + r.w; z3 = z3 > 0.f ? z3 : NEG_SLOPE * z3;
    acc += z0 * a.x + z1 * a.y + z2 * a.z + z3 * a.w;
  }
  eOut[t] = acc;
  atomicMax(maxEnc + (long long)d * H + h, fenc(acc));
}

// ---------------------------------------------------------------------------
// Edge pass 2: ex = exp(e - max[dst]) ; atomic sum per (dst, head)
// ---------------------------------------------------------------------------
__global__ void edge_exp(const float* __restrict__ eIn,
                         const unsigned* __restrict__ maxEnc,
                         const long long* __restrict__ dst,
                         float* __restrict__ alphaNum, float* __restrict__ nodeSum,
                         int E, int H) {
  long long t = (long long)blockIdx.x * blockDim.x + threadIdx.x;
  if (t >= (long long)E * H) return;
  int e = (int)(t / H), h = (int)(t % H);
  long long d = dst[e];
  float m  = fdec(maxEnc[(long long)d * H + h]);
  float ex = expf(eIn[t] - m);
  alphaNum[t] = ex;
  atomicAdd(nodeSum + (long long)d * H + h, ex);
}

// ---------------------------------------------------------------------------
// Edge pass 3: out[dst,h,:] += alpha * xl[src,h,:]   (4 channels / thread)
// ---------------------------------------------------------------------------
__global__ void edge_agg(const float* __restrict__ XL,
                         const float* __restrict__ alphaNum, const float* __restrict__ nodeSum,
                         const long long* __restrict__ src, const long long* __restrict__ dst,
                         float* __restrict__ acc, int E, int H, int C) {
  long long t = (long long)blockIdx.x * blockDim.x + threadIdx.x;
  int c4n = C >> 2;
  if (t >= (long long)E * H * c4n) return;
  int c4 = (int)(t % c4n);
  long long eh = t / c4n;
  int h = (int)(eh % H);
  long long e = eh / H;
  long long s = src[e], d = dst[e];
  float alpha = alphaNum[eh] / (nodeSum[(long long)d * H + h] + 1e-16f);
  float4 v = *(const float4*)(XL + ((long long)s * H + h) * C + (c4 << 2));
  float* o = acc + ((long long)d * H + h) * C + (c4 << 2);
  atomicAdd(o + 0, alpha * v.x);
  atomicAdd(o + 1, alpha * v.y);
  atomicAdd(o + 2, alpha * v.z);
  atomicAdd(o + 3, alpha * v.w);
}

// ---------------------------------------------------------------------------
// Finalize: bias (+ ELU for hidden layers)
// ---------------------------------------------------------------------------
__global__ void finalize_elu(const float* __restrict__ acc, const float* __restrict__ bias,
                             float* __restrict__ out, int N, int D) {
  long long i = (long long)blockIdx.x * blockDim.x + threadIdx.x;
  if (i >= (long long)N * D) return;
  float v = acc[i] + bias[(int)(i % D)];
  out[i] = v > 0.f ? v : expm1f(v);
}
__global__ void finalize_linear(const float* __restrict__ acc, const float* __restrict__ bias,
                                float* __restrict__ out, int N, int D) {
  long long i = (long long)blockIdx.x * blockDim.x + threadIdx.x;
  if (i >= (long long)N * D) return;
  out[i] = acc[i] + bias[(int)(i % D)];
}

// ---------------------------------------------------------------------------
// Host orchestration
// ---------------------------------------------------------------------------
static inline unsigned gridFor(long long n, int blk) {
  return (unsigned)((n + blk - 1) / blk);
}

extern "C" void kernel_launch(void* const* d_in, const int* in_sizes, int n_in,
                              void* d_out, int out_size, void* d_ws, size_t ws_size,
                              hipStream_t stream) {
  (void)in_sizes; (void)n_in; (void)out_size; (void)ws_size;
  const int N = N_NODES, E = N_EDGES;
  const float*     x    = (const float*)d_in[0];
  const long long* eidx = (const long long*)d_in[1];
  const long long* src  = eidx;
  const long long* dst  = eidx + E;

  struct Layer { const float *Wl,*bl,*Wr,*br,*att,*b; int Din, H, C; bool elu; };
  Layer L[4];
  for (int i = 0; i < 4; ++i) {
    const int base = 2 + 6 * i;
    L[i].Wl  = (const float*)d_in[base + 0];
    L[i].bl  = (const float*)d_in[base + 1];
    L[i].Wr  = (const float*)d_in[base + 2];
    L[i].br  = (const float*)d_in[base + 3];
    L[i].att = (const float*)d_in[base + 4];
    L[i].b   = (const float*)d_in[base + 5];
  }
  L[0] = {L[0].Wl, L[0].bl, L[0].Wr, L[0].br, L[0].att, L[0].b, 128, 4, 128, true};
  L[1] = {L[1].Wl, L[1].bl, L[1].Wr, L[1].br, L[1].att, L[1].b, 512, 4, 128, true};
  L[2] = {L[2].Wl, L[2].bl, L[2].Wr, L[2].br, L[2].att, L[2].b, 512, 4, 128, true};
  L[3] = {L[3].Wl, L[3].bl, L[3].Wr, L[3].br, L[3].att, L[3].b, 512, 1,  64, false};

  // ---- carve workspace ----
  char* w = (char*)d_ws;
  auto alloc = [&](size_t bytes) {
    char* p = w;
    w += (bytes + 255) & ~(size_t)255;
    return p;
  };
  const size_t maxD = 512;
  _Float16* Xhi  = (_Float16*)alloc((size_t)N * maxD * 2);
  _Float16* Xlo  = (_Float16*)alloc((size_t)N * maxD * 2);
  _Float16* WLhi = (_Float16*)alloc(maxD * maxD * 2);
  _Float16* WLlo = (_Float16*)alloc(maxD * maxD * 2);
  _Float16* WRhi = (_Float16*)alloc(maxD * maxD * 2);
  _Float16* WRlo = (_Float16*)alloc(maxD * maxD * 2);
  float* XL   = (float*)alloc((size_t)N * maxD * 4);
  float* XR   = (float*)alloc((size_t)N * maxD * 4);
  float* ACC  = (float*)alloc((size_t)N * maxD * 4);
  float* HBUF = (float*)alloc((size_t)N * maxD * 4);
  float* eScore   = (float*)alloc((size_t)E * 4 * 4);
  float* alphaNum = (float*)alloc((size_t)E * 4 * 4);
  unsigned* maxEnc  = (unsigned*)alloc((size_t)N * 4 * 4);
  float*    nodeSum = (float*)alloc((size_t)N * 4 * 4);

  const int T = 256;

  for (int l = 0; l < 4; ++l) {
    const int Din = L[l].Din, H = L[l].H, C = L[l].C, HC = H * C;
    const float* hin = (l == 0) ? x : HBUF;

    // 1) split activations + weights (weights transposed to [HC, Din])
    split_f16<<<gridFor((long long)N * Din, T), T, 0, stream>>>(hin, Xhi, Xlo, (long long)N * Din);
    transpose_split_f16<<<gridFor((long long)Din * HC, T), T, 0, stream>>>(L[l].Wl, WLhi, WLlo, Din, HC);
    transpose_split_f16<<<gridFor((long long)Din * HC, T), T, 0, stream>>>(L[l].Wr, WRhi, WRlo, Din, HC);

    // 2) GEMMs: XL = h@Wl + bl ; XR = h@Wr + br   (WMMA, split-f16, 16x64/wave)
    long long waves = (long long)(N / 16) * (HC / 64);
    unsigned gblk = gridFor(waves, 8);  // 8 waves / 256-thread block
    gemm_split_wmma<<<gblk, T, 0, stream>>>(Xhi, Xlo, WLhi, WLlo, L[l].bl, XL, N, Din, HC);
    gemm_split_wmma<<<gblk, T, 0, stream>>>(Xhi, Xlo, WRhi, WRlo, L[l].br, XR, N, Din, HC);

    // 3) init softmax state + aggregation buffer
    fill_u32<<<gridFor((long long)N * H, T), T, 0, stream>>>(maxEnc, ENC_NEG_INF, (long long)N * H);
    fill_u32<<<gridFor((long long)N * H, T), T, 0, stream>>>((unsigned*)nodeSum, 0u, (long long)N * H);
    fill_u32<<<gridFor((long long)N * HC, T), T, 0, stream>>>((unsigned*)ACC, 0u, (long long)N * HC);

    // 4) segment softmax + aggregation over edges
    edge_score<<<gridFor((long long)E * H, T), T, 0, stream>>>(XL, XR, src, dst, L[l].att,
                                                               eScore, maxEnc, E, H, C);
    edge_exp<<<gridFor((long long)E * H, T), T, 0, stream>>>(eScore, maxEnc, dst,
                                                             alphaNum, nodeSum, E, H);
    edge_agg<<<gridFor((long long)E * H * (C / 4), T), T, 0, stream>>>(XL, alphaNum, nodeSum,
                                                                      src, dst, ACC, E, H, C);

    // 5) bias + activation
    if (L[l].elu)
      finalize_elu<<<gridFor((long long)N * HC, T), T, 0, stream>>>(ACC, L[l].b, HBUF, N, HC);
    else
      finalize_linear<<<gridFor((long long)N * HC, T), T, 0, stream>>>(ACC, L[l].b,
                                                                       (float*)d_out, N, HC);
  }
}
